// LocalExperts_5566277615659
// MI455X (gfx1250) — compile-verified, matrix operand order checked by reading, not run on hardware
//
#include <hip/hip_runtime.h>
#include <hip/hip_bf16.h>

// ---------------------------------------------------------------------------
// Problem constants (from reference): E=8, W=8, C=1024, D=512, F=2048
// Per expert: M = W*C = 8192 token rows.
// out[e] = relu(X[e] @ Wi[e]) @ Wo[e], all fp32.
// Strategy: split-bf16 (hi+lo) x3-product WMMA for fp32-grade accuracy on the
// bf16 matrix pipe (8x MACs/instr vs V_WMMA_F32_16X16X4_F32).
// ---------------------------------------------------------------------------

#define E_    8
#define MROWS 8192   // W*C rows per expert
#define D_    512
#define F_    2048
#define MT    32     // token rows per workgroup
#define FC    128    // F-chunk width

typedef __attribute__((ext_vector_type(16))) __bf16 v16bf;
typedef __attribute__((ext_vector_type(8)))  float  v8f;

union Frag { v16bf v; uint4 q[2]; };

static __device__ inline unsigned short f2bf(float f) {
  unsigned u = __builtin_bit_cast(unsigned, f);
  u += 0x7FFFu + ((u >> 16) & 1u);            // round-to-nearest-even
  return (unsigned short)(u >> 16);
}
static __device__ inline float bf2f(unsigned short h) {
  unsigned u = ((unsigned)h) << 16;
  return __builtin_bit_cast(float, u);
}

// ---------------------------------------------------------------------------
// Pre-pass: transpose [E][R][C] fp32 -> [E][C][R] bf16 hi/lo pair in d_ws.
// Makes every WMMA B-fragment a contiguous 16B load per lane in the hot loop.
// ---------------------------------------------------------------------------
__global__ __launch_bounds__(256) void transpose_split(
    const float* __restrict__ src,
    unsigned short* __restrict__ dhi,
    unsigned short* __restrict__ dlo,
    int R, int C)
{
  __shared__ float tile[32][33];
  const int e = blockIdx.z;
  const float* s = src + (size_t)e * R * C;
  const size_t obase = (size_t)e * R * C;
  const int c0 = blockIdx.x * 32, r0 = blockIdx.y * 32;
  const int tx = threadIdx.x, ty = threadIdx.y;

  for (int j = ty; j < 32; j += 8)
    tile[j][tx] = s[(size_t)(r0 + j) * C + (c0 + tx)];
  __syncthreads();

  for (int j = ty; j < 32; j += 8) {
    float v = tile[tx][j];                       // src[r0+tx][c0+j]
    unsigned short h = f2bf(v);
    size_t o = obase + (size_t)(c0 + j) * R + (r0 + tx);
    dhi[o] = h;
    dlo[o] = f2bf(v - bf2f(h));
  }
}

// ---------------------------------------------------------------------------
// Fused expert FFN. Grid: (MROWS/MT, E). Block: 256 threads = 8 wave32.
// Wave w: GEMM1 covers f-columns [w*16, w*16+16) of each 128-wide F chunk;
//         GEMM2 covers d-columns [w*64, w*64+64) with 8 persistent C tiles.
// Dynamic LDS: X tile bf16 hi/lo (64KB) + H chunk bf16 hi/lo (16KB) = 80KB.
// ---------------------------------------------------------------------------
__global__ __launch_bounds__(256) void moe_ffn_kernel(
    const float* __restrict__ x,
    const unsigned short* __restrict__ wiHi, const unsigned short* __restrict__ wiLo,
    const unsigned short* __restrict__ woHi, const unsigned short* __restrict__ woLo,
    float* __restrict__ out)
{
  extern __shared__ __align__(16) char smem[];
  unsigned short* xHi = (unsigned short*)smem;   // [MT][D_]
  unsigned short* xLo = xHi + MT * D_;
  unsigned short* hHi = xLo + MT * D_;           // [MT][FC]
  unsigned short* hLo = hHi + MT * FC;

  const int e = blockIdx.y;
  const int mBase = blockIdx.x * MT;
  const float* xe   = x   + (size_t)e * MROWS * D_ + (size_t)mBase * D_;
  float*       oute = out + (size_t)e * MROWS * D_ + (size_t)mBase * D_;
  const unsigned short* wih = wiHi + (size_t)e * F_ * D_;  // [F][D]
  const unsigned short* wil = wiLo + (size_t)e * F_ * D_;
  const unsigned short* woh = woHi + (size_t)e * D_ * F_;  // [D][F]
  const unsigned short* wol = woLo + (size_t)e * D_ * F_;

  const int tid   = threadIdx.x;
  const int lane  = tid & 31;
  const int w     = tid >> 5;
  const int lhalf = (lane & 16) ? 8 : 0;   // K sub-offset per lane half
  const int l15   = lane & 15;

  // ---- Stage X tile into LDS as bf16 hi/lo (coalesced float4 loads) ----
  for (int i = tid; i < MT * D_ / 4; i += 256) {
    float4 v = ((const float4*)xe)[i];
    int o = i * 4;
    unsigned short h;
    h = f2bf(v.x); xHi[o + 0] = h; xLo[o + 0] = f2bf(v.x - bf2f(h));
    h = f2bf(v.y); xHi[o + 1] = h; xLo[o + 1] = f2bf(v.y - bf2f(h));
    h = f2bf(v.z); xHi[o + 2] = h; xLo[o + 2] = f2bf(v.z - bf2f(h));
    h = f2bf(v.w); xHi[o + 3] = h; xLo[o + 3] = f2bf(v.w - bf2f(h));
  }
  __syncthreads();

  const v8f zero = {0.f, 0.f, 0.f, 0.f, 0.f, 0.f, 0.f, 0.f};
  v8f acc2[8];                       // [nt*2 + mt] persistent output tiles
#pragma unroll
  for (int i = 0; i < 8; ++i) acc2[i] = zero;

  const int fbW   = w * 16;          // GEMM1 f sub-block inside chunk
  const int dBase = w * 64;          // GEMM2 d block

  for (int ch = 0; ch < F_ / FC; ++ch) {
    const int f0 = ch * FC;
    v8f acc1[2] = {zero, zero};      // H tiles for mt = 0,1

    const unsigned short* wihRow = wih + (size_t)(f0 + fbW + l15) * D_;
    const unsigned short* wilRow = wil + (size_t)(f0 + fbW + l15) * D_;

    // ---- GEMM1: Hc = X_tile @ Wi[:, chunk]  (K = 512, 32 per step) ----
    for (int k = 0; k < D_; k += 32) {
      const int a = k + lhalf;
      Frag bh, bl;
      bh.q[0] = *(const uint4*)(wihRow + a);
      bh.q[1] = *(const uint4*)(wihRow + a + 16);
      bl.q[0] = *(const uint4*)(wilRow + a);
      bl.q[1] = *(const uint4*)(wilRow + a + 16);
#pragma unroll
      for (int mt = 0; mt < 2; ++mt) {
        const unsigned short* xr  = xHi + (mt * 16 + l15) * D_;
        const unsigned short* xrl = xLo + (mt * 16 + l15) * D_;
        Frag ah, al;
        ah.q[0] = *(const uint4*)(xr  + a); ah.q[1] = *(const uint4*)(xr  + a + 16);
        al.q[0] = *(const uint4*)(xrl + a); al.q[1] = *(const uint4*)(xrl + a + 16);
        acc1[mt] = __builtin_amdgcn_wmma_f32_16x16x32_bf16(false, ah.v, false, bh.v, (short)0, acc1[mt], false, false);
        acc1[mt] = __builtin_amdgcn_wmma_f32_16x16x32_bf16(false, ah.v, false, bl.v, (short)0, acc1[mt], false, false);
        acc1[mt] = __builtin_amdgcn_wmma_f32_16x16x32_bf16(false, al.v, false, bh.v, (short)0, acc1[mt], false, false);
      }
    }

    __syncthreads();   // previous chunk's GEMM2 finished reading hHi/hLo

    // ---- ReLU + split -> LDS (C layout: VGPR i = rows i / i+8) ----
#pragma unroll
    for (int mt = 0; mt < 2; ++mt) {
#pragma unroll
      for (int i = 0; i < 8; ++i) {
        float v = acc1[mt][i];
        v = v > 0.f ? v : 0.f;
        int row = mt * 16 + i + ((lane >> 4) << 3);
        int col = fbW + l15;
        unsigned short h = f2bf(v);
        hHi[row * FC + col] = h;
        hLo[row * FC + col] = f2bf(v - bf2f(h));
      }
    }
    __syncthreads();

    // ---- GEMM2: Out += Hc @ Wo[chunk, :]  (K = 128, 32 per step) ----
    for (int kk = 0; kk < FC; kk += 32) {
      const int a = kk + lhalf;
      Frag ah[2], al[2];
#pragma unroll
      for (int mt = 0; mt < 2; ++mt) {
        const unsigned short* hr  = hHi + (mt * 16 + l15) * FC;
        const unsigned short* hrl = hLo + (mt * 16 + l15) * FC;
        ah[mt].q[0] = *(const uint4*)(hr  + a); ah[mt].q[1] = *(const uint4*)(hr  + a + 16);
        al[mt].q[0] = *(const uint4*)(hrl + a); al[mt].q[1] = *(const uint4*)(hrl + a + 16);
      }
      const int ag = f0 + a;
#pragma unroll
      for (int nt = 0; nt < 4; ++nt) {
        const unsigned short* wr  = woh + (size_t)(dBase + nt * 16 + l15) * F_;
        const unsigned short* wrl = wol + (size_t)(dBase + nt * 16 + l15) * F_;
        Frag bh, bl;
        bh.q[0] = *(const uint4*)(wr  + ag); bh.q[1] = *(const uint4*)(wr  + ag + 16);
        bl.q[0] = *(const uint4*)(wrl + ag); bl.q[1] = *(const uint4*)(wrl + ag + 16);
#pragma unroll
        for (int mt = 0; mt < 2; ++mt) {
          v8f c = acc2[nt * 2 + mt];
          c = __builtin_amdgcn_wmma_f32_16x16x32_bf16(false, ah[mt].v, false, bh.v, (short)0, c, false, false);
          c = __builtin_amdgcn_wmma_f32_16x16x32_bf16(false, ah[mt].v, false, bl.v, (short)0, c, false, false);
          c = __builtin_amdgcn_wmma_f32_16x16x32_bf16(false, al[mt].v, false, bh.v, (short)0, c, false, false);
          acc2[nt * 2 + mt] = c;
        }
      }
    }
  }

  // ---- Store fp32 output tiles ----
#pragma unroll
  for (int nt = 0; nt < 4; ++nt)
#pragma unroll
    for (int mt = 0; mt < 2; ++mt)
#pragma unroll
      for (int i = 0; i < 8; ++i) {
        int row = mt * 16 + i + ((lane >> 4) << 3);
        int col = dBase + nt * 16 + l15;
        oute[(size_t)row * D_ + col] = acc2[nt * 2 + mt][i];
      }
}

// ---------------------------------------------------------------------------
// Host launch. Inputs: [0]=x fp32 [E,W,C,D], [1]=capacity (unused by the
// reference math), [2]=wi fp32 [E,D,F], [3]=wo fp32 [E,F,D].
// d_ws layout: wiHi | wiLo | woHi | woLo (bf16, transposed), 64 MB total.
// ---------------------------------------------------------------------------
extern "C" void kernel_launch(void* const* d_in, const int* in_sizes, int n_in,
                              void* d_out, int out_size, void* d_ws, size_t ws_size,
                              hipStream_t stream) {
  (void)in_sizes; (void)n_in; (void)out_size; (void)ws_size;
  const float* x  = (const float*)d_in[0];
  const float* wi = (const float*)d_in[2];
  const float* wo = (const float*)d_in[3];
  float* out = (float*)d_out;

  unsigned short* ws0 = (unsigned short*)d_ws;
  const size_t wsz = (size_t)E_ * F_ * D_;      // 8,388,608 elements each
  unsigned short* wiHi = ws0;
  unsigned short* wiLo = ws0 + wsz;
  unsigned short* woHi = ws0 + 2 * wsz;
  unsigned short* woLo = ws0 + 3 * wsz;

  dim3 tb(32, 8);
  // wi: [E][D][F] -> [E][F][D]
  transpose_split<<<dim3(F_ / 32, D_ / 32, E_), tb, 0, stream>>>(wi, wiHi, wiLo, D_, F_);
  // wo: [E][F][D] -> [E][D][F]
  transpose_split<<<dim3(D_ / 32, F_ / 32, E_), tb, 0, stream>>>(wo, woHi, woLo, F_, D_);

  const size_t shmem = (size_t)(MT * D_ * 2 + MT * FC * 2) * sizeof(unsigned short); // 80 KB
  moe_ffn_kernel<<<dim3(MROWS / MT, E_), 256, shmem, stream>>>(
      x, wiHi, wiLo, woHi, woLo, out);
}
